// Flux2Attention_83236466196676
// MI455X (gfx1250) — compile-verified, matrix operand order checked by reading, not run on hardware
//
#include <hip/hip_runtime.h>
#include <hip/hip_bf16.h>
#include <cstdint>

#define HEADS 24
#define HD 128
#define DMODEL 3072
#define D3 (3 * HEADS * HD)      // 9216
#define S_TXT 512
#define S_IMG 2048
#define S_ALL (S_TXT + S_IMG)    // 2560
#define SCALE 0.08838834764831845f   // 1/sqrt(128)

typedef __attribute__((ext_vector_type(16))) __bf16 v16bf;
typedef __attribute__((ext_vector_type(8)))  float  v8f;

__device__ __forceinline__ v8f wmma_bf16(v16bf a, v16bf b, v8f c) {
  return __builtin_amdgcn_wmma_f32_16x16x32_bf16(
      /*neg_a=*/false, a, /*neg_b=*/false, b,
      /*c_mod=*/(short)0, c, /*reuse_a=*/false, /*reuse_b=*/false);
}

// LDS byte offset of a shared-memory pointer: LDS aperture keeps the byte
// offset in addr[31:0] (ISA 10.2 aperture rules), so truncation is exact.
__device__ __forceinline__ uint32_t lds_off(const void* p) {
  return (uint32_t)(uintptr_t)p;
}

// Async DMA of 32 contiguous bytes global -> LDS (2 x b128, ASYNCcnt += 2).
// offset:16 applies to BOTH the LDS and the global address (ISA §10 async).
__device__ __forceinline__ void async_copy32(uint32_t lds, const void* gptr) {
  asm volatile(
      "global_load_async_to_lds_b128 %0, %1, off\n\t"
      "global_load_async_to_lds_b128 %0, %1, off offset:16"
      :: "v"(lds), "v"(gptr) : "memory");
}

// Fragment loader for 16-bit A/B operands of v_wmma_*_16x16x32:
//   lanes 0-15 : row (lane),    K = {k0+0..7,  k0+16..23}
//   lanes 16-31: row (lane-16), K = {k0+8..15, k0+24..31}
__device__ __forceinline__ v16bf load_frag(const __bf16* base, int ld, int k0, int lane) {
  int r  = lane & 15;
  int ko = (lane >> 4) << 3;          // 0 or 8
  const __bf16* p = base + (size_t)r * ld + k0 + ko;
  union { v16bf v; uint4 q[2]; } u;
  u.q[0] = *reinterpret_cast<const uint4*>(p);
  u.q[1] = *reinterpret_cast<const uint4*>(p + 16);
  return u.v;
}

// ---------------------------------------------------------------- conversions
__global__ void cvt_f32_bf16(const float* __restrict__ in, __bf16* __restrict__ out, size_t n) {
  size_t i = (size_t)blockIdx.x * blockDim.x + threadIdx.x;
  size_t stride = (size_t)gridDim.x * blockDim.x;
  for (; i < n; i += stride) out[i] = (__bf16)in[i];
}

// in: fp32 [K][N] row-major -> out: bf16 [N][K]. LDS-tiled 32x32, both sides coalesced.
__global__ __launch_bounds__(256) void transpose_bf16(const float* __restrict__ in,
                                                      __bf16* __restrict__ out,
                                                      int K, int N) {
  __shared__ __bf16 tile[32][33];
  const int nt = blockIdx.x * 32, kt = blockIdx.y * 32;
  const int tx = threadIdx.x, ty = threadIdx.y;          // 32 x 8
  for (int i = 0; i < 32; i += 8)
    tile[ty + i][tx] = (__bf16)in[(size_t)(kt + ty + i) * N + nt + tx];
  __syncthreads();
  for (int i = 0; i < 32; i += 8)
    out[(size_t)(nt + ty + i) * K + kt + tx] = tile[tx][ty + i];
}

// ---------------------------------------------------------------- WMMA GEMM
// C[M][N] (fp32) = A[M][K](bf16) * Bt[N][K]^T (bf16) + bias.
// 128x128 tile, K-step 32, 8 waves (2M x 4N), double-buffered async LDS staging.
#define G_LDSS 40   // padded LDS row stride (bf16); 80 B keeps b128 alignment

__global__ __launch_bounds__(256) void gemm_bf16_wmma(
    const __bf16* __restrict__ A, const __bf16* __restrict__ Bt,
    const float* __restrict__ bias, float* __restrict__ C,
    int M, int N, int K) {
  __shared__ __bf16 As[2][128 * G_LDSS];
  __shared__ __bf16 Bs[2][128 * G_LDSS];

  const int tid  = threadIdx.x;
  const int lane = tid & 31;
  const int wave = tid >> 5;
  const int wm   = (wave >> 2) * 64;
  const int wn   = (wave & 3) * 32;
  const int bm   = blockIdx.y * 128;
  const int bn   = blockIdx.x * 128;

  v8f acc[4][2];
  for (int i = 0; i < 4; ++i)
    for (int j = 0; j < 2; ++j) acc[i][j] = (v8f){0.f,0.f,0.f,0.f,0.f,0.f,0.f,0.f};

  const int lrow = tid >> 1;            // 0..127
  const int lcol = (tid & 1) << 4;      // 0 or 16
  const __bf16* ga = A  + (size_t)(bm + lrow) * K + lcol;
  const __bf16* gb = Bt + (size_t)(bn + lrow) * K + lcol;
  const uint32_t laA = lds_off(&As[0][lrow * G_LDSS + lcol]);
  const uint32_t laB = lds_off(&Bs[0][lrow * G_LDSS + lcol]);
  const uint32_t bufB = (uint32_t)(128 * G_LDSS * sizeof(__bf16));

  // prologue: stage tile 0 into buffer 0 (ASYNCcnt += 4)
  async_copy32(laA, ga);
  async_copy32(laB, gb);

  int buf = 0;
  for (int k0 = 0; k0 < K; k0 += 32) {
    if (k0 + 32 < K) {
      // stage next tile into the other buffer, then wait only for current tile
      async_copy32(laA + (buf ^ 1) * bufB, ga + k0 + 32);
      async_copy32(laB + (buf ^ 1) * bufB, gb + k0 + 32);
      asm volatile("s_wait_asynccnt 0x4" ::: "memory");
    } else {
      asm volatile("s_wait_asynccnt 0x0" ::: "memory");
    }
    __syncthreads();

    v16bf af[4], bf[2];
    for (int im = 0; im < 4; ++im)
      af[im] = load_frag(&As[buf][(wm + im * 16) * G_LDSS], G_LDSS, 0, lane);
    for (int in = 0; in < 2; ++in)
      bf[in] = load_frag(&Bs[buf][(wn + in * 16) * G_LDSS], G_LDSS, 0, lane);
    for (int im = 0; im < 4; ++im)
      for (int in = 0; in < 2; ++in)
        acc[im][in] = wmma_bf16(af[im], bf[in], acc[im][in]);
    __syncthreads();
    buf ^= 1;
  }

  const int cr = lane & 15;
  const int rh = (lane >> 4) << 3;
  for (int im = 0; im < 4; ++im)
    for (int in = 0; in < 2; ++in) {
      int gm0 = bm + wm + im * 16 + rh;
      int gn  = bn + wn + in * 16 + cr;
      float bv = bias ? bias[gn] : 0.f;
      for (int r = 0; r < 8; ++r)
        C[(size_t)(gm0 + r) * N + gn] = acc[im][in][r] + bv;
    }
}

// ---------------------------------------------------- RMSNorm + RoPE + concat
__global__ void normrope_kernel(
    const float* __restrict__ qkv,   // [2048][9216] (img)
    const float* __restrict__ tqkv,  // [512][9216]  (txt)
    const float* __restrict__ rcos, const float* __restrict__ rsin,
    const float* __restrict__ g_q,  const float* __restrict__ g_k,
    const float* __restrict__ g_aq, const float* __restrict__ g_ak,
    __bf16* __restrict__ Qb, __bf16* __restrict__ Kb, __bf16* __restrict__ Vt) {
  const int s    = blockIdx.x;
  const int h    = blockIdx.y;
  const int part = blockIdx.z;          // 0=q 1=k 2=v
  const int t    = threadIdx.x;         // 0..63 -> pair (2t, 2t+1)
  const bool txt = (s < S_TXT);

  const float* src = txt ? (tqkv + (size_t)s * D3)
                         : (qkv  + (size_t)(s - S_TXT) * D3);
  const int col = part * (HEADS * HD) + h * HD;
  float x0 = src[col + 2 * t];
  float x1 = src[col + 2 * t + 1];

  if (part == 2) {                       // V: convert + transpose only
    __bf16* v = Vt + (size_t)h * HD * S_ALL;
    v[(size_t)(2 * t)     * S_ALL + s] = (__bf16)x0;
    v[(size_t)(2 * t + 1) * S_ALL + s] = (__bf16)x1;
    return;
  }

  float ss = x0 * x0 + x1 * x1;
  for (int off = 16; off; off >>= 1) ss += __shfl_xor(ss, off, 32);
  __shared__ float red[2];
  if ((t & 31) == 0) red[t >> 5] = ss;
  __syncthreads();
  ss = red[0] + red[1];
  float rms = rsqrtf(ss * (1.0f / HD) + 1e-5f);

  const float* g = txt ? (part == 0 ? g_aq : g_ak) : (part == 0 ? g_q : g_k);
  float n0 = x0 * rms * g[2 * t];
  float n1 = x1 * rms * g[2 * t + 1];

  const int rrow = txt ? s : (s - S_TXT);
  float c0 = rcos[rrow * HD + 2 * t],     s0 = rsin[rrow * HD + 2 * t];
  float c1 = rcos[rrow * HD + 2 * t + 1], s1 = rsin[rrow * HD + 2 * t + 1];
  float r0 = n0 * c0 - n1 * s0;          // rot = (-x1, x0)
  float r1 = n1 * c1 + n0 * s1;

  __bf16* dst = (part == 0 ? Qb : Kb) + (size_t)h * S_ALL * HD + (size_t)s * HD;
  dst[2 * t]     = (__bf16)r0;
  dst[2 * t + 1] = (__bf16)r1;
}

// ---------------------------------------------------------------- attention
// Flash-style. grid = (S_ALL/128, HEADS), block = 256 (8 waves).
// KV chunks of 32 staged into double-buffered LDS via async-to-LDS DMA so the
// next chunk's DMA overlaps the current chunk's WMMA + softmax.
#define P_LDSS 40
__global__ __launch_bounds__(256) void attn_wmma_kernel(
    const __bf16* __restrict__ Q, const __bf16* __restrict__ Kk,
    const __bf16* __restrict__ Vt, __bf16* __restrict__ Out) {
  const int head = blockIdx.y;
  const int tid = threadIdx.x, lane = tid & 31, wave = tid >> 5;
  const int qrow = blockIdx.x * 128 + wave * 16;

  const __bf16* Qh = Q  + (size_t)head * S_ALL * HD;
  const __bf16* Kh = Kk + (size_t)head * S_ALL * HD;
  const __bf16* Vh = Vt + (size_t)head * HD * S_ALL;

  __shared__ __bf16 Ks[2][32 * HD];     // K chunk  [32 seq][128 d]   (2 x 8 KB)
  __shared__ __bf16 Vs[2][HD * 32];     // V^T chunk [128 d][32 seq]  (2 x 8 KB)
  __shared__ __bf16 Plds[8][16 * P_LDSS];

  v16bf qf[4];
  for (int ks = 0; ks < 4; ++ks)
    qf[ks] = load_frag(Qh + (size_t)qrow * HD, HD, ks * 32, lane);

  float m[8], l[8];
  v8f o[8];
  for (int r = 0; r < 8; ++r) { m[r] = -1e30f; l[r] = 0.f; }
  for (int d = 0; d < 8; ++d) o[d] = (v8f){0.f,0.f,0.f,0.f,0.f,0.f,0.f,0.f};

  __bf16* P = &Plds[wave][0];
  const int cr = lane & 15;
  const int rh = (lane >> 4) << 3;

  // per-thread staging addresses (buffer 0)
  const uint32_t lk0 = lds_off(&Ks[0][0]) + tid * 32;               // 8 KB linear
  const uint32_t lv0 = lds_off(&Vs[0][0]) + (tid >> 1) * 64 + (tid & 1) * 32;
  const uint32_t kBufB = (uint32_t)(32 * HD * sizeof(__bf16));
  const uint32_t vBufB = (uint32_t)(HD * 32 * sizeof(__bf16));
  const __bf16* gK = Kh + tid * 16;                                 // + kv*HD
  const __bf16* gV = Vh + (size_t)(tid >> 1) * S_ALL + (tid & 1) * 16;  // + kv

  // prologue: stage chunk 0 into buffer 0 (ASYNCcnt += 4)
  async_copy32(lk0, gK);
  async_copy32(lv0, gV);

  int buf = 0;
  for (int kv = 0; kv < S_ALL; kv += 32) {
    if (kv + 32 < S_ALL) {
      async_copy32(lk0 + (buf ^ 1) * kBufB, gK + (size_t)(kv + 32) * HD);
      async_copy32(lv0 + (buf ^ 1) * vBufB, gV + kv + 32);
      asm volatile("s_wait_asynccnt 0x4" ::: "memory");
    } else {
      asm volatile("s_wait_asynccnt 0x0" ::: "memory");
    }
    __syncthreads();

    // S = Q @ K^T from LDS
    v8f sc[2];
    sc[0] = (v8f){0.f,0.f,0.f,0.f,0.f,0.f,0.f,0.f};
    sc[1] = (v8f){0.f,0.f,0.f,0.f,0.f,0.f,0.f,0.f};
    for (int nt = 0; nt < 2; ++nt)
      for (int ks = 0; ks < 4; ++ks) {
        v16bf kf = load_frag(&Ks[buf][nt * 16 * HD], HD, ks * 32, lane);
        sc[nt] = wmma_bf16(qf[ks], kf, sc[nt]);
      }

    // online softmax (row stats live across each 16-lane half)
    float alpha[8];
    for (int r = 0; r < 8; ++r) {
      float a0 = sc[0][r] * SCALE, a1 = sc[1][r] * SCALE;
      float mx = fmaxf(a0, a1);
      for (int off = 8; off; off >>= 1) mx = fmaxf(mx, __shfl_xor(mx, off, 16));
      float nm = fmaxf(m[r], mx);
      alpha[r] = __expf(m[r] - nm);
      m[r] = nm;
      float p0 = __expf(a0 - nm), p1 = __expf(a1 - nm);
      sc[0][r] = p0; sc[1][r] = p1;
      float rs = p0 + p1;
      for (int off = 8; off; off >>= 1) rs += __shfl_xor(rs, off, 16);
      l[r] = l[r] * alpha[r] + rs;
    }
    for (int d = 0; d < 8; ++d)
      for (int r = 0; r < 8; ++r) o[d][r] *= alpha[r];

    // C-layout P -> LDS -> A-layout fragment
    for (int nt = 0; nt < 2; ++nt)
      for (int r = 0; r < 8; ++r)
        P[(r + rh) * P_LDSS + nt * 16 + cr] = (__bf16)sc[nt][r];
    __syncthreads();
    v16bf pf = load_frag(P, P_LDSS, 0, lane);

    // O += P @ V from LDS (V^T rows contiguous over seq)
    for (int d = 0; d < 8; ++d) {
      v16bf vf = load_frag(&Vs[buf][d * 16 * 32], 32, 0, lane);
      o[d] = wmma_bf16(pf, vf, o[d]);
    }
    __syncthreads();   // all waves done with Ks/Vs/P before they are restaged
    buf ^= 1;
  }

  for (int d = 0; d < 8; ++d)
    for (int r = 0; r < 8; ++r) {
      size_t row = qrow + r + rh;
      Out[row * (size_t)(HEADS * HD) + head * HD + d * 16 + cr] =
          (__bf16)(o[d][r] / l[r]);
    }
}

// ---------------------------------------------------------------- launcher
extern "C" void kernel_launch(void* const* d_in, const int* in_sizes, int n_in,
                              void* d_out, int out_size, void* d_ws, size_t ws_size,
                              hipStream_t stream) {
  (void)in_sizes; (void)n_in; (void)out_size; (void)ws_size;
  const float* hs        = (const float*)d_in[0];
  const float* enc       = (const float*)d_in[1];
  const float* rcos      = (const float*)d_in[2];
  const float* rsin      = (const float*)d_in[3];
  const float* w_qkv     = (const float*)d_in[4];
  const float* w_add_kv  = (const float*)d_in[5];
  const float* b_add_kv  = (const float*)d_in[6];
  const float* g_q       = (const float*)d_in[7];
  const float* g_k       = (const float*)d_in[8];
  const float* g_aq      = (const float*)d_in[9];
  const float* g_ak      = (const float*)d_in[10];
  const float* w_out     = (const float*)d_in[11];
  const float* b_out     = (const float*)d_in[12];
  const float* w_add_out = (const float*)d_in[13];
  const float* b_add_out = (const float*)d_in[14];
  float* outp = (float*)d_out;

  char* ws = (char*)d_ws;
  size_t off = 0;
  auto alloc = [&](size_t bytes) -> void* {
    off = (off + 255) & ~(size_t)255;
    void* p = ws + off;
    off += bytes;
    return p;
  };

  __bf16* hs_b   = (__bf16*)alloc((size_t)S_IMG * DMODEL * 2);
  __bf16* enc_b  = (__bf16*)alloc((size_t)S_TXT * DMODEL * 2);
  __bf16* Wstage = (__bf16*)alloc((size_t)D3 * DMODEL * 2);     // reused 4x (stream-ordered)
  float*  qkv    = (float*) alloc((size_t)S_IMG * D3 * 4);
  float*  tqkv   = (float*) alloc((size_t)S_TXT * D3 * 4);
  __bf16* Qb     = (__bf16*)alloc((size_t)HEADS * S_ALL * HD * 2);
  __bf16* Kb     = (__bf16*)alloc((size_t)HEADS * S_ALL * HD * 2);
  __bf16* Vt     = (__bf16*)alloc((size_t)HEADS * HD * S_ALL * 2);
  __bf16* attn   = (__bf16*)alloc((size_t)S_ALL * HEADS * HD * 2);

  // 1) activations -> bf16
  cvt_f32_bf16<<<2048, 256, 0, stream>>>(hs,  hs_b,  (size_t)S_IMG * DMODEL);
  cvt_f32_bf16<<<2048, 256, 0, stream>>>(enc, enc_b, (size_t)S_TXT * DMODEL);

  dim3 tb(32, 8);

  // 2) img QKV GEMM
  transpose_bf16<<<dim3(D3 / 32, DMODEL / 32), tb, 0, stream>>>(w_qkv, Wstage, DMODEL, D3);
  gemm_bf16_wmma<<<dim3(D3 / 128, S_IMG / 128), 256, 0, stream>>>(
      hs_b, Wstage, nullptr, qkv, S_IMG, D3, DMODEL);

  // 3) txt QKV GEMM (+bias), reusing Wstage
  transpose_bf16<<<dim3(D3 / 32, DMODEL / 32), tb, 0, stream>>>(w_add_kv, Wstage, DMODEL, D3);
  gemm_bf16_wmma<<<dim3(D3 / 128, S_TXT / 128), 256, 0, stream>>>(
      enc_b, Wstage, b_add_kv, tqkv, S_TXT, D3, DMODEL);

  // 4) RMSNorm + RoPE + concat into per-head Q/K/V^T bf16
  normrope_kernel<<<dim3(S_ALL, HEADS, 3), 64, 0, stream>>>(
      qkv, tqkv, rcos, rsin, g_q, g_k, g_aq, g_ak, Qb, Kb, Vt);

  // 5) flash attention
  attn_wmma_kernel<<<dim3(S_ALL / 128, HEADS), 256, 0, stream>>>(Qb, Kb, Vt, attn);

  // 6) img out-projection -> d_out[0 : 2048*3072]
  transpose_bf16<<<dim3(DMODEL / 32, DMODEL / 32), tb, 0, stream>>>(w_out, Wstage, DMODEL, DMODEL);
  gemm_bf16_wmma<<<dim3(DMODEL / 128, S_IMG / 128), 256, 0, stream>>>(
      attn + (size_t)S_TXT * DMODEL, Wstage, b_out, outp, S_IMG, DMODEL, DMODEL);

  // 7) txt out-projection -> d_out[2048*3072 : ]
  transpose_bf16<<<dim3(DMODEL / 32, DMODEL / 32), tb, 0, stream>>>(w_add_out, Wstage, DMODEL, DMODEL);
  gemm_bf16_wmma<<<dim3(DMODEL / 128, S_TXT / 128), 256, 0, stream>>>(
      attn, Wstage, b_add_out, outp + (size_t)S_IMG * DMODEL, S_TXT, DMODEL, DMODEL);
}
